// MILC_57612691308927
// MI455X (gfx1250) — matrix operand order repaired and checked by reading, not compiled, not run on hardware
//
#include <hip/hip_runtime.h>

typedef _Float16 f16;
typedef __attribute__((ext_vector_type(16))) _Float16 v16h;
typedef __attribute__((ext_vector_type(8)))  float    v8f;

#define NCOL1 237568   /* 8192*29 */
#define NCOL2 212992   /* 8192*26 */
#define NCOL3 196608   /* 8192*24 */

__device__ __forceinline__ v8f wmma16(v16h a, v16h b, v8f c) {
  // D = A(16x32 f16) * B(32x16 f16) + C(16x16 f32)
  return __builtin_amdgcn_wmma_f32_16x16x32_f16(false, a, false, b, (short)0, c, false, false);
}

// Gather A fragment (16x32) from row-major f16 [m][k] (stride elems). CDNA5 layout:
// lane l (m=l&15, half=l>>4): v0..3 hold K = half*8 + 0..7, v4..7 hold K = 16+half*8+0..7.
__device__ __forceinline__ v16h fragA(const f16* base, int mbase, int koff, int lane, int stride) {
  const int m = mbase + (lane & 15);
  const int half = lane >> 4;
  const f16* row = base + (size_t)m * stride + koff + half * 8;
  v16h a;
#pragma unroll
  for (int i = 0; i < 4; ++i) {
    a[2*i]     = row[2*i];
    a[2*i + 1] = row[2*i + 1];
    a[8 + 2*i]     = row[16 + 2*i];
    a[8 + 2*i + 1] = row[16 + 2*i + 1];
  }
  return a;
}

// Gather B fragment (32x16) from n-major staging [n][k] (stride elems). CDNA5 layout:
// lane l (n=l&15, half=l>>4): v0..3 hold K = half*8 + 0..7, v4..7 hold K = 16+half*8+0..7.
__device__ __forceinline__ v16h fragB(const f16* base, int nbase, int koff, int lane, int stride) {
  const int n = nbase + (lane & 15);
  const int half = lane >> 4;
  const f16* row = base + (size_t)n * stride + koff + half * 8;
  v16h b;
#pragma unroll
  for (int i = 0; i < 8; ++i) { b[i] = row[i]; b[8 + i] = row[16 + i]; }
  return b;
}

struct GP {
  const f16*   A16;    // [Mp][Kp] f16 weights, zero-padded
  const float* bias;   // [M]
  const float* bias2;  // [M] (stage 4 only: bih+bhh)
  const float* Bf32;   // stage 0 source (x)
  const f16*   Bf16;   // stages 1..4 source
  f16*   out16;
  float* out32;
  int M, N, K, Kp;
};

// B element loaders (im2col fused): return B[k][col] as f16.
template<int STAGE>
__device__ __forceinline__ f16 loadB(const GP& p, int k, int c) {
  if constexpr (STAGE == 0) {           // conv1: x[s][ci][t+kk], K=53*4
    int ci = k >> 2, kk = k & 3;
    int s = c / 29, t = c - s * 29;
    return (f16)p.Bf32[(size_t)s * 1696 + ci * 32 + t + kk];
  } else if constexpr (STAGE == 1) {    // conv2: h1[ci][s*29+t+kk], K=64*4
    int ci = k >> 2, kk = k & 3;
    int s = c / 26, t = c - s * 26;
    return p.Bf16[(size_t)ci * NCOL1 + s * 29 + t + kk];
  } else if constexpr (STAGE == 2) {    // conv3: h2[ci][s*26+t+kk], K=128*3
    int ci = k / 3, kk = k - ci * 3;
    int s = c / 24, t = c - s * 24;
    return p.Bf16[(size_t)ci * NCOL2 + s * 26 + t + kk];
  } else if constexpr (STAGE == 3) {    // FC: h3 flat f=ci*24+tt, K=4800
    int ci = k / 24, tt = k - ci * 24;
    return p.Bf16[(size_t)ci * NCOL3 + c * 24 + tt];
  } else {                              // x-gates: enc[d][s], K=256
    return p.Bf16[(size_t)k * 8192 + c];
  }
}

// Tiled WMMA GEMM, BM=128 BN=64 BK=32. A fragments straight from global f16
// (zero-padded, L2 resident). B tile double-buffered in LDS; global gathers for
// chunk k+1 are issued before the WMMAs of chunk k to hide latency.
template<int STAGE>
__global__ void __launch_bounds__(256) gemm_wmma(GP p) {
  __shared__ f16 Bs[2][64 * 32];       // [n][k], n-major
  const int tid = threadIdx.x;
  const int m0 = blockIdx.y * 128, n0 = blockIdx.x * 64;
  const int wid = tid >> 5, lane = tid & 31;
  const int wm = wid & 3, wn = wid >> 2;

  const v8f vz = {0.f, 0.f, 0.f, 0.f, 0.f, 0.f, 0.f, 0.f};
  v8f acc[2][2] = {{vz, vz}, {vz, vz}};

  const int nch = (p.K + 31) >> 5;
  f16 breg[8];
  auto gatherB = [&](int kc) {
    const int k0 = kc * 32;
#pragma unroll
    for (int i = 0; i < 8; ++i) {
      int e = tid + i * 256;
      int n = e >> 5, k = e & 31;
      int gc = n0 + n, gk = k0 + k;
      f16 v = (f16)0.f;
      if (gc < p.N && gk < p.K) v = loadB<STAGE>(p, gk, gc);
      breg[i] = v;
    }
  };
  auto commitB = [&](int buf) {
    f16* d = Bs[buf];
#pragma unroll
    for (int i = 0; i < 8; ++i) d[tid + i * 256] = breg[i];
  };

  gatherB(0);
  commitB(0);
  __syncthreads();

  for (int kc = 0; kc < nch; ++kc) {
    const int nxt = kc + 1;
    if (nxt < nch) gatherB(nxt);                    // overlap with WMMAs below
    const f16* bs = Bs[kc & 1];
    v16h a0 = fragA(p.A16, m0 + wm * 32,      kc * 32, lane, p.Kp);
    v16h a1 = fragA(p.A16, m0 + wm * 32 + 16, kc * 32, lane, p.Kp);
    v16h b0 = fragB(bs, wn * 32,      0, lane, 32);
    v16h b1 = fragB(bs, wn * 32 + 16, 0, lane, 32);
    acc[0][0] = wmma16(a0, b0, acc[0][0]);
    acc[0][1] = wmma16(a0, b1, acc[0][1]);
    acc[1][0] = wmma16(a1, b0, acc[1][0]);
    acc[1][1] = wmma16(a1, b1, acc[1][1]);
    if (nxt < nch) commitB(nxt & 1);
    __syncthreads();
  }

  const int half = lane >> 4, nl = lane & 15;
#pragma unroll
  for (int ti = 0; ti < 2; ++ti)
#pragma unroll
    for (int tj = 0; tj < 2; ++tj)
#pragma unroll
      for (int r = 0; r < 8; ++r) {
        int gm = m0 + (wm * 2 + ti) * 16 + r + 8 * half;
        int gn = n0 + (wn * 2 + tj) * 16 + nl;
        if (gm < p.M && gn < p.N) {
          float v = acc[ti][tj][r] + p.bias[gm];
          if constexpr (STAGE == 4) v += p.bias2[gm];
          if constexpr (STAGE <= 2) v = fmaxf(v, 0.f);
          if constexpr (STAGE == 4) p.out32[(size_t)gm * p.N + gn] = v;
          else                      p.out16[(size_t)gm * p.N + gn] = (f16)v;
        }
      }
}

// f32 -> f16 with zero padding to [Mp][Kp].
__global__ void __launch_bounds__(256) cvt_pad(const float* src, f16* dst,
                                               int M, int K, int Mp, int Kp) {
  int i = blockIdx.x * 256 + threadIdx.x;
  if (i < Mp * Kp) {
    int m = i / Kp, k = i - m * Kp;
    dst[i] = (m < M && k < K) ? (f16)src[(size_t)m * K + k] : (f16)0.f;
  }
}

// Persistent BiLSTM recurrence, one workgroup per direction (blockIdx.x = dir).
// Wave w owns batch tile [16w,16w+16); for each h-block it computes the i/f/g/o
// gate tiles itself, so gates meet in registers and c never leaves VGPRs.
// LDS: h double-buffer (2x32KB) + Whh resident (128KB) = 192KB (CDNA5 WGP: 320KB).
// Whh is preloaded via CDNA5 async global->LDS copies (ASYNCcnt).
__global__ void __launch_bounds__(256) lstm_kernel(
    const float* __restrict__ xg_f, const float* __restrict__ xg_b,
    const f16* __restrict__ whh16_f, const f16* __restrict__ whh16_b,
    float* __restrict__ hf, float* __restrict__ hb) {
  extern __shared__ char smem[];
  f16* hbuf0 = (f16*)smem;                    // [128][128]
  f16* hbuf1 = (f16*)(smem + 32768);          // [128][128]
  f16* whh_lds = (f16*)(smem + 65536);        // [512][128]

  const int dir = blockIdx.x;
  const float* xg  = dir ? xg_b : xg_f;
  const f16*   whh = dir ? whh16_b : whh16_f;
  float*       hs  = dir ? hb : hf;

  const int tid = threadIdx.x, lane = tid & 31, w = tid >> 5;
  const int half = lane >> 4, nl = lane & 15;
  const v8f vz = {0.f, 0.f, 0.f, 0.f, 0.f, 0.f, 0.f, 0.f};

  // Async-copy Whh (512x128 f16 = 128KB) into LDS: 8192 x 16B transfers.
  {
    uint32_t ldsbase = (uint32_t)(uintptr_t)whh_lds;
    for (int i = 0; i < 32; ++i) {
      int idx = tid + i * 256;
      uint32_t dst = ldsbase + idx * 16;
      const f16* src = whh + idx * 8;
      asm volatile("global_load_async_to_lds_b128 %0, %1, off"
                   :: "v"(dst), "v"(src) : "memory");
    }
    asm volatile("s_wait_asynccnt 0x0" ::: "memory");
  }
  // h0 = 0, c = 0 (c persists in registers across all 64 steps)
  for (int i = 0; i < 32; ++i) hbuf0[tid + i * 256] = (f16)0.f;
  v8f creg[8];
#pragma unroll
  for (int i = 0; i < 8; ++i) creg[i] = vz;
  __syncthreads();

  for (int step = 0; step < 64; ++step) {
    const int t = dir ? (63 - step) : step;
    const f16* hc = (step & 1) ? hbuf1 : hbuf0;
    f16*       hn = (step & 1) ? hbuf0 : hbuf1;

    v16h afr[4];
#pragma unroll
    for (int kc = 0; kc < 4; ++kc) afr[kc] = fragA(hc, w * 16, kc * 32, lane, 128);

#pragma unroll
    for (int hbk = 0; hbk < 8; ++hbk) {
      v8f zi = vz, zf = vz, zg = vz, zo = vz;
#pragma unroll
      for (int kc = 0; kc < 4; ++kc) {
        v16h b0 = fragB(whh_lds,   0 + hbk * 16, kc * 32, lane, 128);
        v16h b1 = fragB(whh_lds, 128 + hbk * 16, kc * 32, lane, 128);
        v16h b2 = fragB(whh_lds, 256 + hbk * 16, kc * 32, lane, 128);
        v16h b3 = fragB(whh_lds, 384 + hbk * 16, kc * 32, lane, 128);
        zi = wmma16(afr[kc], b0, zi);
        zf = wmma16(afr[kc], b1, zf);
        zg = wmma16(afr[kc], b2, zg);
        zo = wmma16(afr[kc], b3, zo);
      }
      const int hh = hbk * 16 + nl;
#pragma unroll
      for (int r = 0; r < 8; ++r) {
        int brow = w * 16 + r + 8 * half;
        size_t cb = (size_t)brow * 64 + t;
        float vi = zi[r] + xg[(size_t)(hh      ) * 8192 + cb];
        float vf = zf[r] + xg[(size_t)(hh + 128) * 8192 + cb];
        float vg = zg[r] + xg[(size_t)(hh + 256) * 8192 + cb];
        float vo = zo[r] + xg[(size_t)(hh + 384) * 8192 + cb];
        float cc = creg[hbk][r];
        float si = 1.f / (1.f + expf(-vi));
        float sf = 1.f / (1.f + expf(-vf));
        float so = 1.f / (1.f + expf(-vo));
        cc = sf * cc + si * tanhf(vg);
        float ho = so * tanhf(cc);
        creg[hbk][r] = cc;
        hn[brow * 128 + hh] = (f16)ho;
        hs[((size_t)t * 128 + brow) * 128 + hh] = ho;
      }
    }
    __syncthreads();
  }
}

// Attention pooling + decoder, one workgroup (256 thr) per batch row.
__global__ void __launch_bounds__(256) attn_decode(
    const float* __restrict__ hf, const float* __restrict__ hb,
    const float* __restrict__ wa1, const float* __restrict__ ba1,
    const float* __restrict__ wa2, const float* __restrict__ ba2,
    const float* __restrict__ wd1, const float* __restrict__ bd1,
    const float* __restrict__ wd2, const float* __restrict__ bd2,
    float* __restrict__ out) {
  __shared__ float s1[64 * 128];
  __shared__ float sc[64];
  __shared__ float pooled[256];
  __shared__ float d1[256];
  const int b = blockIdx.x, tid = threadIdx.x;

  const float* hfl = hf + ((size_t)63 * 128 + b) * 128;   // last timestep
  const float* hbl = hb + ((size_t)63 * 128 + b) * 128;
  for (int i = 0; i < 32; ++i) {
    int p = tid + i * 256;
    int t = p >> 7, j = p & 127;
    const float* hft = hf + ((size_t)t * 128 + b) * 128;
    const float* hbt = hb + ((size_t)t * 128 + b) * 128;
    const float* wr = wa1 + j * 512;
    float acc = ba1[j];
    for (int q = 0; q < 128; ++q) acc += hft[q] * wr[q];
    for (int q = 0; q < 128; ++q) acc += hbt[q] * wr[128 + q];
    for (int q = 0; q < 128; ++q) acc += hfl[q] * wr[256 + q];
    for (int q = 0; q < 128; ++q) acc += hbl[q] * wr[384 + q];
    s1[p] = fmaxf(acc, 0.f);
  }
  __syncthreads();
  if (tid < 64) {
    float acc = ba2[0];
    for (int j = 0; j < 128; ++j) acc += s1[tid * 128 + j] * wa2[j];
    sc[tid] = acc;
  }
  __syncthreads();
  if (tid == 0) {
    float mx = sc[0];
    for (int t2 = 1; t2 < 64; ++t2) mx = fmaxf(mx, sc[t2]);
    float sum = 0.f;
    for (int t2 = 0; t2 < 64; ++t2) { float e = expf(sc[t2] - mx); sc[t2] = e; sum += e; }
    float inv = 1.f / sum;
    for (int t2 = 0; t2 < 64; ++t2) sc[t2] *= inv;
  }
  __syncthreads();
  {
    int h = tid;
    const float* base = (h < 128) ? (hf + (size_t)b * 128 + h) : (hb + (size_t)b * 128 + (h - 128));
    float acc = 0.f;
    for (int t2 = 0; t2 < 64; ++t2) acc += sc[t2] * base[(size_t)t2 * 16384];
    pooled[h] = acc;
  }
  __syncthreads();
  {
    int j = tid;
    const float* wr = wd1 + j * 256;
    float acc = bd1[j];
    for (int h = 0; h < 256; ++h) acc += pooled[h] * wr[h];
    d1[j] = fmaxf(acc, 0.f);
  }
  __syncthreads();
  if (tid < 2) {
    const float* wr = wd2 + tid * 256;
    float acc = bd2[tid];
    for (int h = 0; h < 256; ++h) acc += d1[h] * wr[h];
    out[b * 2 + tid] = acc;
  }
}

extern "C" void kernel_launch(void* const* d_in, const int* in_sizes, int n_in,
                              void* d_out, int out_size, void* d_ws, size_t ws_size,
                              hipStream_t stream) {
  (void)in_sizes; (void)n_in; (void)out_size; (void)ws_size;
  const float* x    = (const float*)d_in[0];
  const float* cw1  = (const float*)d_in[1];
  const float* cb1  = (const float*)d_in[2];
  const float* cw2  = (const float*)d_in[3];
  const float* cb2  = (const float*)d_in[4];
  const float* cw3  = (const float*)d_in[5];
  const float* cb3  = (const float*)d_in[6];
  const float* wfc  = (const float*)d_in[7];
  const float* bfc  = (const float*)d_in[8];
  const float* wihf = (const float*)d_in[9];
  const float* whhf = (const float*)d_in[10];
  const float* bihf = (const float*)d_in[11];
  const float* bhhf = (const float*)d_in[12];
  const float* wihb = (const float*)d_in[13];
  const float* whhb = (const float*)d_in[14];
  const float* bihb = (const float*)d_in[15];
  const float* bhhb = (const float*)d_in[16];
  const float* wa1  = (const float*)d_in[17];
  const float* ba1  = (const float*)d_in[18];
  const float* wa2  = (const float*)d_in[19];
  const float* ba2  = (const float*)d_in[20];
  const float* wd1  = (const float*)d_in[21];
  const float* bd1  = (const float*)d_in[22];
  const float* wd2  = (const float*)d_in[23];
  const float* bd2  = (const float*)d_in[24];

  char* ws = (char*)d_ws;
  size_t off = 0;
  auto take = [&](size_t bytes) -> void* {
    void* p = ws + off;
    off = (off + bytes + 255) & ~(size_t)255;
    return p;
  };
  f16*   h1     = (f16*)take((size_t)64  * NCOL1 * 2);
  f16*   h2     = (f16*)take((size_t)128 * NCOL2 * 2);
  f16*   h3     = (f16*)take((size_t)200 * NCOL3 * 2);
  f16*   enc    = (f16*)take((size_t)256 * 8192 * 2);
  float* xgf    = (float*)take((size_t)512 * 8192 * 4);
  float* xgb    = (float*)take((size_t)512 * 8192 * 4);
  float* hf     = (float*)take((size_t)64 * 128 * 128 * 4);
  float* hb     = (float*)take((size_t)64 * 128 * 128 * 4);
  // zero-padded f16 weight copies
  f16*   a1w    = (f16*)take((size_t)128 * 224 * 2);
  f16*   a2w    = (f16*)take((size_t)128 * 256 * 2);
  f16*   a3w    = (f16*)take((size_t)256 * 384 * 2);
  f16*   afc    = (f16*)take((size_t)256 * 4800 * 2);
  f16*   aihf   = (f16*)take((size_t)512 * 256 * 2);
  f16*   aihb   = (f16*)take((size_t)512 * 256 * 2);
  f16*   whh16f = (f16*)take((size_t)512 * 128 * 2);
  f16*   whh16b = (f16*)take((size_t)512 * 128 * 2);

  auto cvt = [&](const float* s, f16* d, int M, int K, int Mp, int Kp) {
    int total = Mp * Kp;
    cvt_pad<<<dim3((total + 255) / 256), 256, 0, stream>>>(s, d, M, K, Mp, Kp);
  };
  cvt(cw1,  a1w,    64, 212, 128, 224);
  cvt(cw2,  a2w,   128, 256, 128, 256);
  cvt(cw3,  a3w,   200, 384, 256, 384);
  cvt(wfc,  afc,   256, 4800, 256, 4800);
  cvt(wihf, aihf,  512, 256, 512, 256);
  cvt(wihb, aihb,  512, 256, 512, 256);
  cvt(whhf, whh16f, 512, 128, 512, 128);
  cvt(whhb, whh16b, 512, 128, 512, 128);

  GP p0 = {a1w, cb1, nullptr, x, nullptr, h1, nullptr, 64, NCOL1, 212, 224};
  gemm_wmma<0><<<dim3(NCOL1 / 64, 1), 256, 0, stream>>>(p0);

  GP p1 = {a2w, cb2, nullptr, nullptr, h1, h2, nullptr, 128, NCOL2, 256, 256};
  gemm_wmma<1><<<dim3(NCOL2 / 64, 1), 256, 0, stream>>>(p1);

  GP p2 = {a3w, cb3, nullptr, nullptr, h2, h3, nullptr, 200, NCOL3, 384, 384};
  gemm_wmma<2><<<dim3(NCOL3 / 64, 2), 256, 0, stream>>>(p2);

  GP p3 = {afc, bfc, nullptr, nullptr, h3, enc, nullptr, 256, 8192, 4800, 4800};
  gemm_wmma<3><<<dim3(8192 / 64, 2), 256, 0, stream>>>(p3);

  GP p4f = {aihf, bihf, bhhf, nullptr, enc, nullptr, xgf, 512, 8192, 256, 256};
  gemm_wmma<4><<<dim3(8192 / 64, 4), 256, 0, stream>>>(p4f);
  GP p4b = {aihb, bihb, bhhb, nullptr, enc, nullptr, xgb, 512, 8192, 256, 256};
  gemm_wmma<4><<<dim3(8192 / 64, 4), 256, 0, stream>>>(p4b);

  lstm_kernel<<<dim3(2), 256, 196608, stream>>>(xgf, xgb, whh16f, whh16b, hf, hb);

  attn_decode<<<dim3(128), 256, 0, stream>>>(hf, hb, wa1, ba1, wa2, ba2,
                                             wd1, bd1, wd2, bd2, (float*)d_out);
}
